// AttentionAugmentation_81269371175281
// MI455X (gfx1250) — matrix operand order repaired and checked by reading, not compiled, not run on hardware
//
#include <hip/hip_runtime.h>

typedef __attribute__((ext_vector_type(16))) _Float16 v16h;
typedef __attribute__((ext_vector_type(2)))  _Float16 v2h;
typedef __attribute__((ext_vector_type(8)))  float    v8f;

#define B_   8
#define H_   32
#define W_   32
#define DK_  256
#define DV_  256
#define NH_  8
#define CH_  768
#define DKH  32
#define DVH  32
#define QBLK 128
#define NWAVE 8

static __device__ __forceinline__ v8f wmma16(v16h a, v16h b, v8f c) {
  // D = A(16x32 f16) * B(32x16 f16) + C(16x16 f32)
  return __builtin_amdgcn_wmma_f32_16x16x32_f16(false, a, false, b, (short)0, c, false, false);
}

// B-operand fragment. Tile stored in LDS as T[N][K] with 32 columns.
// Lane layout (wave32, 16-bit B 32x16): N = lane%16, K = (v<4?0:16) + 8*(lane/16) + 2*(v&3) + {0,1}
static __device__ __forceinline__ v16h load_bfrag(const _Float16* t, int lm, int half) {
  v16h r;
#pragma unroll
  for (int v = 0; v < 8; ++v) {
    v2h p = *(const v2h*)(t + lm * 32 + ((v < 4) ? 0 : 16) + 8 * half + 2 * (v & 3));
    r[2 * v]     = p[0];
    r[2 * v + 1] = p[1];
  }
  return r;
}

// A-operand fragment. Tile stored in LDS as T[M][K] with 32 columns.
// Lane layout (ISA 16-bit A 16x32): M = lane%16, K = (v<4?0:16) + 8*(lane/16) + 2*(v&3) + {0,1}
static __device__ __forceinline__ v16h load_afrag(const _Float16* t, int lm, int half) {
  v16h r;
#pragma unroll
  for (int v = 0; v < 8; ++v) {
    int k = ((v < 4) ? 0 : 16) + 8 * half + 2 * (v & 3);
    v2h p = *(const v2h*)(t + lm * 32 + k);
    r[2 * v]     = p[0];
    r[2 * v + 1] = p[1];
  }
  return r;
}

__global__ __launch_bounds__(256)
void attn_aug_kernel(const float* __restrict__ inp,
                     const float* __restrict__ relw,
                     const float* __restrict__ relh,
                     float* __restrict__ out) {
  extern __shared__ char smem_raw[];
  _Float16* relwH = (_Float16*)smem_raw;        // [64][32] (row 63 zero-padded)
  _Float16* relhH = relwH + 64 * 32;            // [64][32]
  _Float16* Klds  = relhH + 64 * 32;            // [32 key][32 d]
  _Float16* VTlds = Klds + 32 * 32;             // [32 dv][32 key]  (transposed V)
  _Float16* Plds  = VTlds + 32 * 32;            // [8 wave][16 M][32 K]
  _Float16* RWlds = Plds + NWAVE * 16 * 32;     // [128 q][64 m]
  _Float16* RHlds = RWlds + 128 * 64;           // [128 q][64 m]

  const int tid  = threadIdx.x;
  const int wave = tid >> 5;
  const int lane = tid & 31;
  const int half = lane >> 4;
  const int lm   = lane & 15;

  const int bid  = blockIdx.x;
  const int qblk = bid & 7;     // 8 query blocks of 128
  const int bn   = bid >> 3;
  const int b    = bn >> 3;     // NH_ == 8
  const int n    = bn & 7;

  const float scale = 0.17677669529663687f;  // (dk/Nh)^-0.5 = 1/sqrt(32)

  // ---- cooperative load of relative-key tables into LDS (f16, zero-padded row 63) ----
  for (int i = tid; i < 64 * 32; i += 256) {
    int m = i >> 5, d = i & 31;
    relwH[i] = (m < 63) ? (_Float16)relw[m * 32 + d] : (_Float16)0.0f;
    relhH[i] = (m < 63) ? (_Float16)relh[m * 32 + d] : (_Float16)0.0f;
  }

  // ---- load this wave's Q tile directly into WMMA A layout (pre-scaled) ----
  const int qg = qblk * QBLK + wave * 16 + lm;  // this lane's query row (M = lm)
  const int qy = qg >> 5, qx = qg & 31;
  const float* qptr = inp + ((size_t)(b * H_ + qy) * W_ + qx) * CH_ + n * DKH;
  v16h qfrag;
#pragma unroll
  for (int v = 0; v < 8; ++v) {
    int k = ((v < 4) ? 0 : 16) + 8 * half + 2 * (v & 3);
    qfrag[2 * v]     = (_Float16)(qptr[k]     * scale);
    qfrag[2 * v + 1] = (_Float16)(qptr[k + 1] * scale);
  }

  __syncthreads();

  // ---- relative logits via WMMA: RW[q][m] = q . relw[m], RH[q][m] = q . relh[m] ----
#pragma unroll
  for (int t = 0; t < 4; ++t) {
    v8f aw = {}; v8f ah = {};
    v16h bw = load_bfrag(relwH + t * 16 * 32, lm, half);
    v16h bh = load_bfrag(relhH + t * 16 * 32, lm, half);
    aw = wmma16(qfrag, bw, aw);
    ah = wmma16(qfrag, bh, ah);
#pragma unroll
    for (int r = 0; r < 8; ++r) {
      int row = wave * 16 + r + 8 * half;     // C layout: M = r + 8*half, N = lm
      RWlds[row * 64 + t * 16 + lm] = (_Float16)aw[r];
      RHlds[row * 64 + t * 16 + lm] = (_Float16)ah[r];
    }
  }

  // ---- flash-attention accumulators ----
  v8f o0 = {}; v8f o1 = {};
  float mrun[8], lrun[8];
#pragma unroll
  for (int r = 0; r < 8; ++r) { mrun[r] = -3.0e38f; lrun[r] = 0.0f; }

  for (int c = 0; c < 32; ++c) {   // key chunk = image row y2 = c (32 keys)
    // cooperative K/V chunk load: 32 keys x 32 dims each, f32 -> f16
    {
      int i  = tid * 4;            // 256 threads * 4 elems = 1024
      int ky = i >> 5;             // key x2
      int d0 = i & 31;             // 4 consecutive dims
      const float* kp = inp + ((size_t)(b * H_ + c) * W_ + ky) * CH_ + DK_     + n * DKH + d0;
      const float* vp = inp + ((size_t)(b * H_ + c) * W_ + ky) * CH_ + 2 * DK_ + n * DVH + d0;
      float4 kv = *(const float4*)kp;
      float4 vv = *(const float4*)vp;
      Klds[ky * 32 + d0 + 0] = (_Float16)kv.x;
      Klds[ky * 32 + d0 + 1] = (_Float16)kv.y;
      Klds[ky * 32 + d0 + 2] = (_Float16)kv.z;
      Klds[ky * 32 + d0 + 3] = (_Float16)kv.w;
      VTlds[(d0 + 0) * 32 + ky] = (_Float16)vv.x;
      VTlds[(d0 + 1) * 32 + ky] = (_Float16)vv.y;
      VTlds[(d0 + 2) * 32 + ky] = (_Float16)vv.z;
      VTlds[(d0 + 3) * 32 + ky] = (_Float16)vv.w;
    }
    __syncthreads();

    // S = Q * K^T for two 16-key tiles
    v8f s0 = {}; v8f s1 = {};
    v16h kb0 = load_bfrag(Klds,           lm, half);
    v16h kb1 = load_bfrag(Klds + 16 * 32, lm, half);
    s0 = wmma16(qfrag, kb0, s0);
    s1 = wmma16(qfrag, kb1, s1);

    // read V fragments now so K/V LDS may be overwritten next iteration
    v16h vb0 = load_bfrag(VTlds,           lm, half);
    v16h vb1 = load_bfrag(VTlds + 16 * 32, lm, half);

    // bias add + online softmax (C layout rows: M = r + 8*half across 16 lanes)
#pragma unroll
    for (int r = 0; r < 8; ++r) {
      int M   = r + 8 * half;
      int row = wave * 16 + M;
      int qg2 = qblk * QBLK + row;
      int y   = qg2 >> 5, x = qg2 & 31;
      float bh  = (float)RHlds[row * 64 + (c + 31 - y)];
      float s0r = s0[r] + bh + (float)RWlds[row * 64 + (lm      + 31 - x)];
      float s1r = s1[r] + bh + (float)RWlds[row * 64 + (lm + 16 + 31 - x)];
      float mx = fmaxf(s0r, s1r);
#pragma unroll
      for (int off = 1; off < 16; off <<= 1) mx = fmaxf(mx, __shfl_xor(mx, off, 32));
      float mnew  = fmaxf(mrun[r], mx);
      float alpha = __expf(mrun[r] - mnew);
      mrun[r] = mnew;
      float p0 = __expf(s0r - mnew);
      float p1 = __expf(s1r - mnew);
      float rs = p0 + p1;
#pragma unroll
      for (int off = 1; off < 16; off <<= 1) rs += __shfl_xor(rs, off, 32);
      lrun[r] = lrun[r] * alpha + rs;
      o0[r] *= alpha;
      o1[r] *= alpha;
      Plds[wave * 512 + M * 32 + lm]      = (_Float16)p0;   // P[M][key 0..15]
      Plds[wave * 512 + M * 32 + 16 + lm] = (_Float16)p1;   // P[M][key 16..31]
    }
    __syncthreads();

    // O += P * V  (K = 32 keys, two dv tiles of 16)
    v16h pfrag = load_afrag(Plds + wave * 512, lm, half);
    o0 = wmma16(pfrag, vb0, o0);
    o1 = wmma16(pfrag, vb1, o1);
  }

  // ---- normalize and store: out[b,y,x, n*32 + dv], fp32 ----
#pragma unroll
  for (int r = 0; r < 8; ++r) {
    int M   = r + 8 * half;
    int qg2 = qblk * QBLK + wave * 16 + M;
    int y   = qg2 >> 5, x = qg2 & 31;
    float inv = 1.0f / lrun[r];
    float* op = out + ((size_t)(b * H_ + y) * W_ + x) * DV_ + n * DVH;
    op[lm]      = o0[r] * inv;
    op[lm + 16] = o1[r] * inv;
  }
}

extern "C" void kernel_launch(void* const* d_in, const int* in_sizes, int n_in,
                              void* d_out, int out_size, void* d_ws, size_t ws_size,
                              hipStream_t stream) {
  const float* inp  = (const float*)d_in[0];
  const float* relw = (const float*)d_in[1];
  const float* relh = (const float*)d_in[2];
  float* out = (float*)d_out;
  (void)in_sizes; (void)n_in; (void)out_size; (void)d_ws; (void)ws_size;

  // LDS: relw(2048) + relh(2048) + K(1024) + VT(1024) + P(4096) + RW(8192) + RH(8192) f16
  size_t smem = (size_t)(2048 + 2048 + 1024 + 1024 + 4096 + 8192 + 8192) * sizeof(_Float16); // 53248 B

  dim3 grid(B_ * NH_ * (H_ * W_ / QBLK));  // 8 * 8 * 8 = 512
  dim3 block(256);                          // 8 waves (wave32)
  hipLaunchKernelGGL(attn_aug_kernel, grid, block, smem, stream, inp, relw, relh, out);
}
